// UOTHead_43276090474729
// MI455X (gfx1250) — matrix operand order repaired and checked by reading, not compiled, not run on hardware
//
#include <hip/hip_runtime.h>
#include <hip/hip_bf16.h>

// ---------------------------------------------------------------------------
// Unbalanced Sinkhorn OT head for MI455X (gfx1250, wave32).
// Pipeline:
//   1) normalize_f16   : L2-normalize features, emit fp16          (32 MB ws)
//   2) gemm_exp (WMMA) : K = exp((f1.f2^T - 1)/eps), fp16          (256 MB ws)
//      - software-pipelined k-loop (double-buffered fragments)
//      - raw v_exp_f32 epilogue (fp16 store flushes tiny values anyway)
//   3) 10x { colsum->b update, rowsum->a update }  (HBM-bound streaming)
//   4) final_corr      : corr = a*K*b^T @ p2 / rowsum, weights = rowsum
//   5) rigid           : weighted Kabsch via 3x3 Jacobi SVD (1 block/batch)
// ---------------------------------------------------------------------------

#define B_ 8
#define N_ 4096
#define M_ 4096
#define C_ 256

#define POWER_        0.9433962264f     /* gamma/(gamma+eps) = 0.5/0.53 */
#define INV_EPS_L2E   48.0898346f       /* (1/0.03) * log2(e)           */
#define PROB1_        (1.0f/4096.0f)
#define PROB2_        (1.0f/4096.0f)

typedef __attribute__((ext_vector_type(16))) _Float16 v16h;
typedef __attribute__((ext_vector_type(8)))  float    v8f;

union H16 { v16h v; unsigned u[8]; };

__device__ __forceinline__ float waveReduce(float v) {
#pragma unroll
  for (int off = 16; off > 0; off >>= 1) v += __shfl_xor(v, off, 32);
  return v;
}

// -------------------------------------------------------------- normalize ---
__global__ __launch_bounds__(256) void normalize_f16_kernel(
    const float* __restrict__ src, _Float16* __restrict__ dst, int rows) {
  int wid  = blockIdx.x * 8 + (int)(threadIdx.x >> 5);
  int lane = (int)(threadIdx.x & 31);
  if (wid >= rows) return;
  const float4* s = (const float4*)(src + (size_t)wid * C_);
  float4 x0 = s[lane];
  float4 x1 = s[lane + 32];
  float ss = x0.x*x0.x + x0.y*x0.y + x0.z*x0.z + x0.w*x0.w
           + x1.x*x1.x + x1.y*x1.y + x1.z*x1.z + x1.w*x1.w;
  ss = waveReduce(ss);
  float inv = rsqrtf(ss + 1e-8f);
  union { _Float16 h[4]; uint2 u; } p0, p1;
  p0.h[0] = (_Float16)(x0.x*inv); p0.h[1] = (_Float16)(x0.y*inv);
  p0.h[2] = (_Float16)(x0.z*inv); p0.h[3] = (_Float16)(x0.w*inv);
  p1.h[0] = (_Float16)(x1.x*inv); p1.h[1] = (_Float16)(x1.y*inv);
  p1.h[2] = (_Float16)(x1.z*inv); p1.h[3] = (_Float16)(x1.w*inv);
  uint2* d = (uint2*)(dst + (size_t)wid * C_);
  d[lane]      = p0.u;
  d[lane + 32] = p1.u;
}

// ------------------------------------------------------------------- init ---
__global__ void init_ab_kernel(float* __restrict__ a, float* __restrict__ b) {
  int i = blockIdx.x * blockDim.x + threadIdx.x;
  if (i < B_ * N_) a[i] = PROB1_;
  if (i < B_ * M_) b[i] = PROB2_;
}

// ------------------------------------------------- WMMA GEMM + exp -> K ----
// One wave computes a 16(n) x 64(m) strip: 4 accumulators, k-loop 256/32,
// fully unrolled with a double-buffered fragment pipeline so the next
// k-step's global_load_b128s are in flight while the 4 WMMAs execute.
// Per-lane operand gathers follow the documented wave32 16-bit layouts:
//   A 16x32: lanes 0-15 row M=lane (K 0..7,16..23), lanes 16-31 (K 8..15,24..31)
//   B 32x16: lanes 0-15 col N=lane (K 0..7,16..23), lanes 16-31 (K 8..15,24..31)
__device__ __forceinline__ void load_fragA(H16& d, const _Float16* A, int k0, int half) {
#pragma unroll
  for (int j = 0; j < 8; ++j) {
    int kk = (j < 4) ? (half * 8 + 2 * j) : (16 + half * 8 + 2 * (j - 4));
    d.u[j] = *(const unsigned*)(A + k0 + kk);
  }
}

__device__ __forceinline__ void load_fragB(H16& d, const _Float16* Bt, int k0, int half) {
#pragma unroll
  for (int j = 0; j < 8; ++j) {
    int kk = (j < 4) ? (half * 8 + 2 * j) : (16 + half * 8 + 2 * (j - 4));
    d.u[j] = *(const unsigned*)(Bt + k0 + kk);
  }
}

__global__ __launch_bounds__(256) void gemm_exp_kernel(
    const _Float16* __restrict__ f1h, const _Float16* __restrict__ f2h,
    _Float16* __restrict__ Km) {
  int wid  = blockIdx.x * 8 + (int)(threadIdx.x >> 5);
  int lane = (int)(threadIdx.x & 31);
  const int mstrips = M_ / 64;                 // 64
  const int tiles_per_b = (N_ / 16) * mstrips; // 16384
  int b   = wid / tiles_per_b;
  int rem = wid - b * tiles_per_b;
  int nt  = rem / mstrips;
  int ms  = rem - nt * mstrips;

  int row  = lane & 15;
  int half = lane >> 4;

  const _Float16* A  = f1h + ((size_t)b * N_ + (size_t)nt * 16 + row) * C_;
  const _Float16* Bp = f2h + ((size_t)b * M_ + (size_t)ms * 64 + row) * C_;

  v8f z = {};
  v8f acc[4]; acc[0] = z; acc[1] = z; acc[2] = z; acc[3] = z;

  H16 av[2];
  H16 bv[2][4];
  load_fragA(av[0], A, 0, half);
#pragma unroll
  for (int t = 0; t < 4; ++t)
    load_fragB(bv[0][t], Bp + (size_t)t * 16 * C_, 0, half);

#pragma unroll
  for (int ks = 0; ks < 8; ++ks) {           // 8 k-steps of 32
    const int cur = ks & 1;
    const int nxt = cur ^ 1;
    if (ks < 7) {                            // prefetch next fragments
      load_fragA(av[nxt], A, (ks + 1) * 32, half);
#pragma unroll
      for (int t = 0; t < 4; ++t)
        load_fragB(bv[nxt][t], Bp + (size_t)t * 16 * C_, (ks + 1) * 32, half);
    }
#pragma unroll
    for (int t = 0; t < 4; ++t)
      acc[t] = __builtin_amdgcn_wmma_f32_16x16x32_f16(
          false, av[cur].v, false, bv[cur][t].v, (short)0, acc[t], false, false);
  }

  // C/D layout: VGPR r -> (M = r + 8*halfsel, N = lane&15)
  _Float16* outb = Km + (size_t)b * N_ * M_;
#pragma unroll
  for (int t = 0; t < 4; ++t) {
#pragma unroll
    for (int r = 0; r < 8; ++r) {
      float cv = acc[t][r];
      // raw v_exp_f32: inputs < -126 would produce denormal/0, which the
      // fp16 store flushes to 0 anyway -> libm-safe expansion unnecessary.
      float kv = __builtin_amdgcn_exp2f((cv - 1.0f) * INV_EPS_L2E);
      int orow = nt * 16 + half * 8 + r;
      int ocol = ms * 64 + t * 16 + row;
      outb[(size_t)orow * M_ + ocol] = (_Float16)kv;
    }
  }
}

// --------------------------------------------- column pass: KTa -> b ------
__global__ __launch_bounds__(256) void colsum_update_b_kernel(
    const _Float16* __restrict__ Km, const float* __restrict__ avec,
    float* __restrict__ bvec) {
  int b = blockIdx.y;
  int m = blockIdx.x * 256 + threadIdx.x;
  const _Float16* Kc = Km + (size_t)b * N_ * M_ + m;
  const float* a = avec + (size_t)b * N_;
  float acc = 0.0f;
  for (int n = 0; n < N_; n += 8) {
    __builtin_prefetch(Kc + (size_t)(n + 32) * M_, 0, 1);
#pragma unroll
    for (int j = 0; j < 8; ++j)
      acc = fmaf((float)Kc[(size_t)(n + j) * M_], a[n + j], acc);
  }
  bvec[(size_t)b * M_ + m] = __powf(PROB2_ / (acc + 1e-8f), POWER_);
}

// ------------------------------------------------ row pass: Kb -> a -------
__global__ __launch_bounds__(256) void rowsum_update_a_kernel(
    const _Float16* __restrict__ Km, const float* __restrict__ bvec,
    float* __restrict__ avec) {
  int b    = blockIdx.y;
  int n    = blockIdx.x * 8 + (int)(threadIdx.x >> 5);
  int lane = (int)(threadIdx.x & 31);
  const _Float16* Kr = Km + ((size_t)b * N_ + n) * M_;
  const float* bv = bvec + (size_t)b * M_;
  float acc = 0.0f;
  for (int base = lane * 8; base < M_; base += 256) {
    __builtin_prefetch(Kr + base + 1024, 0, 1);
#pragma unroll
    for (int j = 0; j < 8; ++j)
      acc = fmaf((float)Kr[base + j], bv[base + j], acc);
  }
  acc = waveReduce(acc);
  if (lane == 0)
    avec[(size_t)b * N_ + n] = __powf(PROB1_ / (acc + 1e-8f), POWER_);
}

// --------------------------------- final: corr = T@p2/rowsum, weights -----
__global__ __launch_bounds__(256) void final_corr_kernel(
    const _Float16* __restrict__ Km, const float* __restrict__ bvec,
    const float* __restrict__ avec, const float* __restrict__ p2,
    float* __restrict__ corr, float* __restrict__ wts) {
  int b    = blockIdx.y;
  int n    = blockIdx.x * 8 + (int)(threadIdx.x >> 5);
  int lane = (int)(threadIdx.x & 31);
  const _Float16* Kr = Km + ((size_t)b * N_ + n) * M_;
  const float* bv = bvec + (size_t)b * M_;
  const float* P2 = p2 + (size_t)b * M_ * 3;
  float s0 = 0.f, sx = 0.f, sy = 0.f, sz = 0.f;
  for (int base = lane * 8; base < M_; base += 256) {
    __builtin_prefetch(Kr + base + 1024, 0, 1);
#pragma unroll
    for (int j = 0; j < 8; ++j) {
      int m = base + j;
      float kb = (float)Kr[m] * bv[m];
      s0 += kb;
      sx = fmaf(kb, P2[m * 3 + 0], sx);
      sy = fmaf(kb, P2[m * 3 + 1], sy);
      sz = fmaf(kb, P2[m * 3 + 2], sz);
    }
  }
#pragma unroll
  for (int off = 16; off > 0; off >>= 1) {
    s0 += __shfl_xor(s0, off, 32);
    sx += __shfl_xor(sx, off, 32);
    sy += __shfl_xor(sy, off, 32);
    sz += __shfl_xor(sz, off, 32);
  }
  if (lane == 0) {
    float a   = avec[(size_t)b * N_ + n];
    float rs  = a * s0;                      // row_sum of T
    float inv = 1.0f / (rs + 1e-8f);
    size_t o  = ((size_t)b * N_ + n) * 3;
    corr[o + 0] = a * sx * inv;
    corr[o + 1] = a * sy * inv;
    corr[o + 2] = a * sz * inv;
    wts[(size_t)b * N_ + n] = rs;
  }
}

// -------------------------------------------- rigid transform (Kabsch) ----
__device__ void jacobi3(float G[3][3], float V[3][3]) {
#pragma unroll
  for (int i = 0; i < 3; ++i)
#pragma unroll
    for (int j = 0; j < 3; ++j) V[i][j] = (i == j) ? 1.0f : 0.0f;
  for (int sweep = 0; sweep < 12; ++sweep) {
    for (int pq = 0; pq < 3; ++pq) {
      int p = (pq == 0) ? 0 : (pq == 1) ? 0 : 1;
      int q = (pq == 0) ? 1 : (pq == 1) ? 2 : 2;
      float apq = G[p][q];
      if (fabsf(apq) < 1e-15f) continue;
      float theta = (G[q][q] - G[p][p]) / (2.0f * apq);
      float tt = copysignf(1.0f, theta) / (fabsf(theta) + sqrtf(theta * theta + 1.0f));
      float c  = 1.0f / sqrtf(tt * tt + 1.0f);
      float s  = tt * c;
      for (int k = 0; k < 3; ++k) {
        float gkp = G[k][p], gkq = G[k][q];
        G[k][p] = c * gkp - s * gkq;
        G[k][q] = s * gkp + c * gkq;
      }
      for (int k = 0; k < 3; ++k) {
        float gpk = G[p][k], gqk = G[q][k];
        G[p][k] = c * gpk - s * gqk;
        G[q][k] = s * gpk + c * gqk;
      }
      for (int k = 0; k < 3; ++k) {
        float vkp = V[k][p], vkq = V[k][q];
        V[k][p] = c * vkp - s * vkq;
        V[k][q] = s * vkp + c * vkq;
      }
    }
  }
}

__global__ __launch_bounds__(256) void rigid_kernel(
    const float* __restrict__ p1, const float* __restrict__ corr,
    const float* __restrict__ w, float* __restrict__ out) {
  __shared__ float red[16][256];
  int b = blockIdx.x, tid = threadIdx.x;
  float s[16];
#pragma unroll
  for (int i = 0; i < 16; ++i) s[i] = 0.0f;

  const float* P1 = p1   + (size_t)b * N_ * 3;
  const float* CB = corr + (size_t)b * N_ * 3;
  const float* W  = w    + (size_t)b * N_;
  for (int n = tid; n < N_; n += 256) {
    float wn = W[n];
    float ax = P1[n*3+0], ay = P1[n*3+1], az = P1[n*3+2];
    float bx = CB[n*3+0], by = CB[n*3+1], bz = CB[n*3+2];
    s[0] += wn;
    s[1] += wn*ax; s[2] += wn*ay; s[3] += wn*az;
    s[4] += wn*bx; s[5] += wn*by; s[6] += wn*bz;
    s[7]  += wn*ax*bx; s[8]  += wn*ax*by; s[9]  += wn*ax*bz;
    s[10] += wn*ay*bx; s[11] += wn*ay*by; s[12] += wn*ay*bz;
    s[13] += wn*az*bx; s[14] += wn*az*by; s[15] += wn*az*bz;
  }
#pragma unroll
  for (int i = 0; i < 16; ++i) red[i][tid] = s[i];
  __syncthreads();
  for (int off = 128; off > 0; off >>= 1) {
    if (tid < off)
#pragma unroll
      for (int i = 0; i < 16; ++i) red[i][tid] += red[i][tid + off];
    __syncthreads();
  }
  if (tid != 0) return;

  float Sw = red[0][0];
  float t  = 1.0f / (Sw + 1e-5f);
  float ca[3], cb[3];
  for (int i = 0; i < 3; ++i) { ca[i] = red[1+i][0] * t; cb[i] = red[4+i][0] * t; }
  float ssum = Sw * t;
  float cov[3][3];
  for (int i = 0; i < 3; ++i)
    for (int j = 0; j < 3; ++j)
      cov[i][j] = red[7 + i*3 + j][0] * t - ca[i] * cb[j] * (2.0f - ssum);

  // SVD via eigen-decomp of cov^T cov
  float G[3][3];
  for (int i = 0; i < 3; ++i)
    for (int j = 0; j < 3; ++j)
      G[i][j] = cov[0][i]*cov[0][j] + cov[1][i]*cov[1][j] + cov[2][i]*cov[2][j];
  float V[3][3];
  jacobi3(G, V);
  float ev[3] = { G[0][0], G[1][1], G[2][2] };
  int idx[3] = { 0, 1, 2 };
  if (ev[idx[0]] < ev[idx[1]]) { int k = idx[0]; idx[0] = idx[1]; idx[1] = k; }
  if (ev[idx[0]] < ev[idx[2]]) { int k = idx[0]; idx[0] = idx[2]; idx[2] = k; }
  if (ev[idx[1]] < ev[idx[2]]) { int k = idx[1]; idx[1] = idx[2]; idx[2] = k; }
  float Vs[3][3], U[3][3];
  for (int j = 0; j < 3; ++j) {
    float sv = sqrtf(fmaxf(ev[idx[j]], 0.0f));
    for (int i = 0; i < 3; ++i) Vs[i][j] = V[i][idx[j]];
    float ux = cov[0][0]*Vs[0][j] + cov[0][1]*Vs[1][j] + cov[0][2]*Vs[2][j];
    float uy = cov[1][0]*Vs[0][j] + cov[1][1]*Vs[1][j] + cov[1][2]*Vs[2][j];
    float uz = cov[2][0]*Vs[0][j] + cov[2][1]*Vs[1][j] + cov[2][2]*Vs[2][j];
    float inv = 1.0f / fmaxf(sv, 1e-12f);
    U[0][j] = ux * inv; U[1][j] = uy * inv; U[2][j] = uz * inv;
  }
  float R[3][3];
  for (int i = 0; i < 3; ++i)
    for (int j = 0; j < 3; ++j)
      R[i][j] = Vs[i][0]*U[j][0] + Vs[i][1]*U[j][1] + Vs[i][2]*U[j][2];
  float det = R[0][0]*(R[1][1]*R[2][2]-R[1][2]*R[2][1])
            - R[0][1]*(R[1][0]*R[2][2]-R[1][2]*R[2][0])
            + R[0][2]*(R[1][0]*R[2][1]-R[1][1]*R[2][0]);
  if (det <= 0.0f) {
    for (int i = 0; i < 3; ++i) Vs[i][2] = -Vs[i][2];
    for (int i = 0; i < 3; ++i)
      for (int j = 0; j < 3; ++j)
        R[i][j] = Vs[i][0]*U[j][0] + Vs[i][1]*U[j][1] + Vs[i][2]*U[j][2];
  }
  for (int i = 0; i < 3; ++i) {
    float tr = cb[i] - (R[i][0]*ca[0] + R[i][1]*ca[1] + R[i][2]*ca[2]);
    out[b*12 + i*4 + 0] = R[i][0];
    out[b*12 + i*4 + 1] = R[i][1];
    out[b*12 + i*4 + 2] = R[i][2];
    out[b*12 + i*4 + 3] = tr;
  }
}

// ------------------------------------------------------------------ host ---
extern "C" void kernel_launch(void* const* d_in, const int* in_sizes, int n_in,
                              void* d_out, int out_size, void* d_ws, size_t ws_size,
                              hipStream_t stream) {
  const float* f1 = (const float*)d_in[0];
  const float* f2 = (const float*)d_in[1];
  const float* p1 = (const float*)d_in[2];
  const float* p2 = (const float*)d_in[3];
  float* out = (float*)d_out;

  char* ws = (char*)d_ws;
  _Float16* f1h = (_Float16*)(ws);                      // 16 MB
  _Float16* f2h = (_Float16*)(ws + ((size_t)16 << 20)); // 16 MB
  _Float16* Km  = (_Float16*)(ws + ((size_t)32 << 20)); // 256 MB
  float* avec = (float*)(ws + ((size_t)288 << 20));     // 128 KB
  float* bvec = avec + (size_t)B_ * N_;                 // 128 KB
  float* wts  = bvec + (size_t)B_ * M_;                 // 128 KB
  float* corr = wts  + (size_t)B_ * N_;                 // 384 KB

  normalize_f16_kernel<<<(B_ * N_) / 8, 256, 0, stream>>>(f1, f1h, B_ * N_);
  normalize_f16_kernel<<<(B_ * M_) / 8, 256, 0, stream>>>(f2, f2h, B_ * M_);
  init_ab_kernel<<<(B_ * N_ + 255) / 256, 256, 0, stream>>>(avec, bvec);

  gemm_exp_kernel<<<(B_ * (N_ / 16) * (M_ / 64)) / 8, 256, 0, stream>>>(f1h, f2h, Km);

  for (int it = 0; it < 10; ++it) {
    colsum_update_b_kernel<<<dim3(M_ / 256, B_), 256, 0, stream>>>(Km, avec, bvec);
    rowsum_update_a_kernel<<<dim3(N_ / 8, B_), 256, 0, stream>>>(Km, bvec, avec);
  }

  final_corr_kernel<<<dim3(N_ / 8, B_), 256, 0, stream>>>(Km, bvec, avec, p2, corr, wts);
  rigid_kernel<<<B_, 256, 0, stream>>>(p1, corr, wts, out);
}
